// SqueezedGRUTF_18631568130457
// MI455X (gfx1250) — matrix operand order repaired and checked by reading, not compiled, not run on hardware
//
#include <hip/hip_runtime.h>
#include <hip/hip_bf16.h>

// ---------------------------------------------------------------------------
// SqueezedGRU (TF GRUCell, reset_after=True) for MI455X / gfx1250.
// bf16 WMMA 16x16x32 (fp32 accum) for all GEMMs; fp32 gate math + state.
// Persistent grid-synced scan kernel with register-resident recurrent weights
// (128 VGPRs/wave of B fragments) and 4-way WMMA ILP on the serial path.
// ---------------------------------------------------------------------------

typedef __bf16 bf16x16 __attribute__((ext_vector_type(16)));
typedef float f32x8  __attribute__((ext_vector_type(8)));

#define B_   32
#define T_   1000
#define H_   512
#define G_   8
#define BT_  (B_ * T_)
#define H3_  (3 * H_)
#define SCAN_WGS 32

#define WMMA_BF16(a, b, c) \
  __builtin_amdgcn_wmma_f32_16x16x32_bf16(false, (a), false, (b), (short)0, (c), false, false)

static __device__ __forceinline__ unsigned short f2bf(float f) {
  unsigned u = __float_as_uint(f);
  u += 0x7FFF + ((u >> 16) & 1u);          // round-to-nearest-even
  return (unsigned short)(u >> 16);
}

// A fragment: 16x32 bf16, row-major source [M][K], lda in elements.
// Lane layout (ISA 7.12.2): m = lane&15, k in {half*8+0..7} ∪ {16+half*8+0..7}.
static __device__ __forceinline__ bf16x16 fragA(const unsigned short* A, int lda) {
  int lane = threadIdx.x & 31;
  int m = lane & 15, half = lane >> 4;
  const unsigned short* r = A + m * lda + half * 8;
  bf16x16 f;
  ((uint4*)&f)[0] = *(const uint4*)(r);
  ((uint4*)&f)[1] = *(const uint4*)(r + 16);
  return f;
}

// B fragment: 32x16 bf16 from TRANSPOSED weights BT[N][K], ldk in elements.
// Lane layout: n = lane&15, k in {half*8+0..7} ∪ {16+half*8+0..7} (matches fragA).
static __device__ __forceinline__ bf16x16 fragBT(const unsigned short* BT, int ldk) {
  int lane = threadIdx.x & 31;
  int n = lane & 15, half = lane >> 4;
  const unsigned short* r = BT + n * ldk + half * 8;
  bf16x16 f;
  ((uint4*)&f)[0] = *(const uint4*)(r);
  ((uint4*)&f)[1] = *(const uint4*)(r + 16);
  return f;
}

// ------------------------------- prep kernels ------------------------------

__global__ void k_cvt_bf16(const float* __restrict__ in, unsigned short* __restrict__ out, int n) {
  int i = blockIdx.x * blockDim.x + threadIdx.x;
  if (i < n) out[i] = f2bf(in[i]);
}

// in: [batch][rows][cols] fp32 -> out: [batch][cols][rows] bf16   (batch = blockIdx.y)
__global__ void k_transpose_bf16(const float* __restrict__ in, unsigned short* __restrict__ out,
                                 int rows, int cols) {
  long base = (long)blockIdx.y * rows * cols;
  int i = blockIdx.x * blockDim.x + threadIdx.x;
  if (i < rows * cols) {
    int r = i / cols, c = i % cols;
    out[base + (long)c * rows + r] = f2bf(in[base + i]);
  }
}

__global__ void k_init(const float* __restrict__ state, float* __restrict__ hf,
                       unsigned short* __restrict__ hb, unsigned* __restrict__ ctr) {
  int i = blockIdx.x * blockDim.x + threadIdx.x;
  if (i < B_ * H_) { float v = state[i]; hf[i] = v; hb[i] = f2bf(v); }
  if (i == 0) *ctr = 0u;
}

// ------------------------- grouped linear-in + ReLU ------------------------
__global__ void __launch_bounds__(256) k_gemm_in(const unsigned short* __restrict__ xbf,
                                                 const unsigned short* __restrict__ wiT,
                                                 float* __restrict__ hin_f,
                                                 unsigned short* __restrict__ hin_bf) {
  int wid  = threadIdx.x >> 5;
  int rows = (blockIdx.x * 8 + wid) * 16;
  int g    = blockIdx.y;
  const unsigned short* A0 = xbf + (long)rows * H_ + g * 64;   // lda = 512
  f32x8 acc[4] = {};
#pragma unroll
  for (int kk = 0; kk < 64; kk += 32) {
    bf16x16 a = fragA(A0 + kk, H_);
#pragma unroll
    for (int nt = 0; nt < 4; ++nt) {
      bf16x16 b = fragBT(wiT + g * 64 * 64 + nt * 16 * 64 + kk, 64);
      acc[nt] = WMMA_BF16(a, b, acc[nt]);
    }
  }
  int lane = threadIdx.x & 31, n = lane & 15, half = lane >> 4;
#pragma unroll
  for (int nt = 0; nt < 4; ++nt) {
    int col = g * 64 + nt * 16 + n;
#pragma unroll
    for (int r = 0; r < 8; ++r) {
      float v = fmaxf(acc[nt][r], 0.f);
      long idx = (long)(rows + r + 8 * half) * H_ + col;
      hin_f[idx]  = v;
      hin_bf[idx] = f2bf(v);
    }
  }
}

// ------------------------- xz = h_in @ kernel + b_ih -----------------------
__global__ void __launch_bounds__(256) k_gemm_xz(const unsigned short* __restrict__ hin_bf,
                                                 const unsigned short* __restrict__ kT,
                                                 const float* __restrict__ b_ih,
                                                 float* __restrict__ xz) {
  int wid  = threadIdx.x >> 5;
  int rows = (blockIdx.x * 8 + wid) * 16;
  int c0   = blockIdx.y * 64;
  const unsigned short* A0 = hin_bf + (long)rows * H_;
  f32x8 acc[4] = {};
  for (int k = 0; k < H_; k += 32) {
    __builtin_prefetch(A0 + k + 64, 0, 1);                     // global_prefetch_b8
    bf16x16 a = fragA(A0 + k, H_);
#pragma unroll
    for (int nt = 0; nt < 4; ++nt) {
      bf16x16 b = fragBT(kT + (long)(c0 + nt * 16) * H_ + k, H_);
      acc[nt] = WMMA_BF16(a, b, acc[nt]);
    }
  }
  int lane = threadIdx.x & 31, n = lane & 15, half = lane >> 4;
#pragma unroll
  for (int nt = 0; nt < 4; ++nt) {
    int col = c0 + nt * 16 + n;
    float bv = b_ih[col];
#pragma unroll
    for (int r = 0; r < 8; ++r)
      xz[(long)(rows + r + 8 * half) * H3_ + col] = acc[nt][r] + bv;
  }
}

// ------------------------------ recurrent scan -----------------------------
// 32 persistent WGs x 256 threads; WG owns 16 h-columns (z/r/n tiles, K=512).
// Waves 0..5: (gate q, M-tile mt) with recurrent weights held in 128 VGPRs.
// Waves 6..7: prefetch this step's xz / h_in rows into L2/L0.
__global__ void __launch_bounds__(256) k_scan(const float* __restrict__ xz,
                                              const unsigned short* __restrict__ rkT,
                                              const float* __restrict__ b_hh,
                                              const float* __restrict__ hin_f,
                                              float* hf0, float* hf1,
                                              unsigned short* hb0, unsigned short* hb1,
                                              unsigned short* __restrict__ yp_bf,
                                              float* __restrict__ state_out,
                                              unsigned* __restrict__ ctr, int nblocks) {
  __shared__ float rec_s[3][32][16];
  int tid = threadIdx.x;
  int c0  = blockIdx.x * 16;
  int wid = tid >> 5, lane = tid & 31;
  int q = wid >> 1, mt = wid & 1;                 // waves 0..5: gate q, M-tile mt

  // Recurrent-weight fragments live in registers for the whole scan.
  bf16x16 breg[16];
  if (wid < 6) {
    const unsigned short* Bq = rkT + (long)(q * H_ + c0) * H_;
#pragma unroll
    for (int kk = 0; kk < 16; ++kk)
      breg[kk] = fragBT(Bq + kk * 32, H_);
  }

  const float*          hf_c = hf0;  float*          hf_n = hf1;
  const unsigned short* hb_c = hb0;  unsigned short* hb_n = hb1;

  for (int t = 0; t < T_; ++t) {
    if (wid < 6) {
      const unsigned short* A0 = hb_c + mt * 16 * H_;
      f32x8 s0 = {}, s1 = {}, s2 = {}, s3 = {};   // 4-way WMMA ILP
#pragma unroll
      for (int kk = 0; kk < 16; kk += 4) {
        s0 = WMMA_BF16(fragA(A0 + (kk + 0) * 32, H_), breg[kk + 0], s0);
        s1 = WMMA_BF16(fragA(A0 + (kk + 1) * 32, H_), breg[kk + 1], s1);
        s2 = WMMA_BF16(fragA(A0 + (kk + 2) * 32, H_), breg[kk + 2], s2);
        s3 = WMMA_BF16(fragA(A0 + (kk + 3) * 32, H_), breg[kk + 3], s3);
      }
      f32x8 acc = (s0 + s1) + (s2 + s3);
      int n = lane & 15, half = lane >> 4;
#pragma unroll
      for (int r = 0; r < 8; ++r)
        rec_s[q][mt * 16 + r + 8 * half][n] = acc[r];
    } else {
      // Prefetch gate-math operands for this step while WMMA waves compute.
      int l = (wid - 6) * 32 + lane;              // 0..63
      int b = l & 31, qq = l >> 5;                // qq in {0,1}
      __builtin_prefetch(xz + (long)(b * T_ + t) * H3_ + qq * H_ + c0, 0, 1);
      if (qq == 0) {
        __builtin_prefetch(xz + (long)(b * T_ + t) * H3_ + 2 * H_ + c0, 0, 1);
        __builtin_prefetch(hin_f + (long)(b * T_ + t) * H_ + c0, 0, 1);
      }
    }
    __syncthreads();

#pragma unroll
    for (int u = 0; u < 2; ++u) {                 // 512 (b,j) pairs over 256 threads
      int idx = tid + u * 256;
      int b = idx >> 4, j = idx & 15;
      int col = c0 + j;
      const float* xr = xz + (long)(b * T_ + t) * H3_;
      float rz = rec_s[0][b][j] + b_hh[col];
      float rr = rec_s[1][b][j] + b_hh[H_ + col];
      float rn = rec_s[2][b][j] + b_hh[2 * H_ + col];
      float z  = 1.f / (1.f + __expf(-(xr[col] + rz)));
      float r  = 1.f / (1.f + __expf(-(xr[H_ + col] + rr)));
      float nn = tanhf(xr[2 * H_ + col] + r * rn);
      float hp = hf_c[b * H_ + col];
      float hnew = z * hp + (1.f - z) * nn;
      hf_n[b * H_ + col] = hnew;
      hb_n[b * H_ + col] = f2bf(hnew);
      long orow = (long)(b * T_ + t) * H_ + col;
      yp_bf[orow] = f2bf(hnew + hin_f[orow]);     // GRU skip, bf16 feed to linear_out
      if (t == T_ - 1) state_out[b * H_ + col] = hnew;
    }
    __syncthreads();

    __threadfence();                              // make h writes device-visible
    if (tid == 0) {
      __hip_atomic_fetch_add(ctr, 1u, __ATOMIC_RELEASE, __HIP_MEMORY_SCOPE_AGENT);
      unsigned tgt = (unsigned)nblocks * (unsigned)(t + 1);
      while (__hip_atomic_load(ctr, __ATOMIC_ACQUIRE, __HIP_MEMORY_SCOPE_AGENT) < tgt)
        __builtin_amdgcn_s_sleep(1);
    }
    __syncthreads();
    __threadfence();                              // acquire other WGs' h for next step

    const float* tf = hf_c; hf_c = hf_n; hf_n = (float*)tf;
    const unsigned short* tb = hb_c; hb_c = hb_n; hb_n = (unsigned short*)tb;
  }
}

// ------------------------- grouped linear-out + ReLU -----------------------
__global__ void __launch_bounds__(256) k_gemm_out(const unsigned short* __restrict__ yp_bf,
                                                  const unsigned short* __restrict__ woT,
                                                  float* __restrict__ y) {
  int wid  = threadIdx.x >> 5;
  int rows = (blockIdx.x * 8 + wid) * 16;
  int g    = blockIdx.y;
  const unsigned short* A0 = yp_bf + (long)rows * H_ + g * 64;
  f32x8 acc[4] = {};
#pragma unroll
  for (int kk = 0; kk < 64; kk += 32) {
    bf16x16 a = fragA(A0 + kk, H_);
#pragma unroll
    for (int nt = 0; nt < 4; ++nt) {
      bf16x16 b = fragBT(woT + g * 64 * 64 + nt * 16 * 64 + kk, 64);
      acc[nt] = WMMA_BF16(a, b, acc[nt]);
    }
  }
  int lane = threadIdx.x & 31, n = lane & 15, half = lane >> 4;
#pragma unroll
  for (int nt = 0; nt < 4; ++nt) {
    int col = g * 64 + nt * 16 + n;
#pragma unroll
    for (int r = 0; r < 8; ++r)
      y[(long)(rows + r + 8 * half) * H_ + col] = fmaxf(acc[nt][r], 0.f);
  }
}

// --------------------------------- launch ----------------------------------

extern "C" void kernel_launch(void* const* d_in, const int* in_sizes, int n_in,
                              void* d_out, int out_size, void* d_ws, size_t ws_size,
                              hipStream_t stream) {
  (void)in_sizes; (void)n_in; (void)out_size; (void)ws_size;
  const float* x     = (const float*)d_in[0];
  const float* state = (const float*)d_in[1];
  const float* w_in  = (const float*)d_in[2];
  const float* kern  = (const float*)d_in[3];
  const float* rkern = (const float*)d_in[4];
  const float* bias  = (const float*)d_in[5];
  const float* w_out = (const float*)d_in[6];

  char* ws = (char*)d_ws;
  size_t o = 0;
  auto alloc = [&](size_t bytes) { size_t r = o; o += (bytes + 255) & ~(size_t)255; return r; };
  unsigned short* xbf  = (unsigned short*)(ws + alloc((size_t)BT_ * H_ * 2));
  float*          hinf = (float*)         (ws + alloc((size_t)BT_ * H_ * 4));
  unsigned short* hinb = (unsigned short*)(ws + alloc((size_t)BT_ * H_ * 2));
  float*          xz   = (float*)         (ws + alloc((size_t)BT_ * H3_ * 4));
  unsigned short* kT   = (unsigned short*)(ws + alloc((size_t)H3_ * H_ * 2));
  unsigned short* rkT  = (unsigned short*)(ws + alloc((size_t)H3_ * H_ * 2));
  unsigned short* wiT  = (unsigned short*)(ws + alloc((size_t)G_ * 64 * 64 * 2));
  unsigned short* woT  = (unsigned short*)(ws + alloc((size_t)G_ * 64 * 64 * 2));
  unsigned short* ypbf = (unsigned short*)(ws + alloc((size_t)BT_ * H_ * 2));
  float*          hf0  = (float*)         (ws + alloc((size_t)B_ * H_ * 4));
  float*          hf1  = (float*)         (ws + alloc((size_t)B_ * H_ * 4));
  unsigned short* hb0  = (unsigned short*)(ws + alloc((size_t)B_ * H_ * 2));
  unsigned short* hb1  = (unsigned short*)(ws + alloc((size_t)B_ * H_ * 2));
  unsigned*       ctr  = (unsigned*)      (ws + alloc(256));

  float* y_out     = (float*)d_out;
  float* state_out = y_out + (size_t)BT_ * H_;

  k_cvt_bf16<<<(BT_ * H_ + 255) / 256, 256, 0, stream>>>(x, xbf, BT_ * H_);
  k_transpose_bf16<<<dim3((H_ * H3_ + 255) / 256, 1), 256, 0, stream>>>(kern, kT, H_, H3_);
  k_transpose_bf16<<<dim3((H_ * H3_ + 255) / 256, 1), 256, 0, stream>>>(rkern, rkT, H_, H3_);
  k_transpose_bf16<<<dim3((64 * 64 + 255) / 256, G_), 256, 0, stream>>>(w_in, wiT, 64, 64);
  k_transpose_bf16<<<dim3((64 * 64 + 255) / 256, G_), 256, 0, stream>>>(w_out, woT, 64, 64);
  k_init<<<(B_ * H_ + 255) / 256, 256, 0, stream>>>(state, hf0, hb0, ctr);

  k_gemm_in<<<dim3(BT_ / 16 / 8, G_), 256, 0, stream>>>(xbf, wiT, hinf, hinb);
  k_gemm_xz<<<dim3(BT_ / 16 / 8, H3_ / 64), 256, 0, stream>>>(hinb, kT, bias, xz);
  k_scan<<<SCAN_WGS, 256, 0, stream>>>(xz, rkT, bias + H3_, hinf,
                                       hf0, hf1, hb0, hb1, ypbf, state_out, ctr, SCAN_WGS);
  k_gemm_out<<<dim3(BT_ / 16 / 8, G_), 256, 0, stream>>>(ypbf, woT, y_out);
}
